// GFNet_78417512890604
// MI455X (gfx1250) — matrix operand order repaired and checked by reading, not compiled, not run on hardware
//
#include <hip/hip_runtime.h>
#include <hip/hip_bf16.h>
#include <math.h>
#include <stdint.h>

typedef __bf16 bf16_t;
typedef bf16_t v16bf __attribute__((ext_vector_type(16)));
typedef float  v8f   __attribute__((ext_vector_type(8)));

#define DIMC 768
#define HIDC 1536
#define NTOK 16384   /* B(64) * 256 tokens */

// ---------------------------------------------------------------------------
// GEMM: C[M,N] = A[M,K] (bf16, row-major) * W[N,K]^T (bf16, row-major [N][K])
// Block 256 thr = 8 waves (2x4 wave grid). Block tile 128(M)x256(N), K-step 32.
// Wave tile 64x64 -> 4x4 wmma accumulators (16 WMMAs per 8 fragment loads).
// Tiles are streamed global->LDS with GLOBAL_LOAD_ASYNC_TO_LDS_B128 (ASYNCcnt),
// double-buffered so tile k+1 DMA overlaps the WMMAs of tile k with zero
// VGPR staging (fixes the scratch spills seen with register staging).
// EPI 0: out_f32 = C + bias[n] + pos[(m%256)*768+n]        (patch embed -> t)
// EPI 1: out_bf16 = gelu(C + bias[n])                      (fc1 -> hidden)
// EPI 2: out_f32[m,n] += C + bias[n]                       (fc2 -> residual t)
// ---------------------------------------------------------------------------
#define BM 128
#define BN 256
#define BK 32
#define LDK 40   /* padded LDS row stride (elements), 80B = 16B-aligned */

union FragBF { v16bf v; uint4 q[2]; };

__device__ __forceinline__ void GFNet_async_b128(unsigned lds_byte_off,
                                                 uint64_t gaddr) {
  // LDS[VDST] <= MEM[VADDR], 16 bytes per lane, tracked by ASYNCcnt.
  asm volatile("global_load_async_to_lds_b128 %0, %1, off"
               :: "v"(lds_byte_off), "v"(gaddr) : "memory");
}

template <int EPI>
__global__ __launch_bounds__(256)
void GFNet_gemm(const bf16_t* __restrict__ A, const bf16_t* __restrict__ W,
                const float* __restrict__ bias, const float* __restrict__ pos,
                float* __restrict__ outF, bf16_t* __restrict__ outH,
                int M, int N, int K) {
  __shared__ __align__(16) bf16_t sA[2][BM * LDK];
  __shared__ __align__(16) bf16_t sB[2][BN * LDK];

  const int t    = threadIdx.x;
  const int wid  = t >> 5;
  const int lane = t & 31;
  const int half = lane >> 4;       // K-half selector for fragments
  const int mrow = lane & 15;

  const int bm0 = blockIdx.x * BM;
  const int bn0 = blockIdx.y * BN;
  const int wm0 = (wid >> 2) * 64;  // wave grid 2 (M) x 4 (N)
  const int wn0 = (wid & 3) * 64;

  v8f acc[4][4] = {};

  // tile loader mapping: 16B chunks (8 bf16). A: 512 chunks (2/thr), B: 1024 (4/thr).
  const int ar0 = t >> 2,          ac0 = (t & 3) * 8;
  const int ar1 = (t + 256) >> 2,  ac1 = ((t + 256) & 3) * 8;

  // global source addresses for this thread's 6 chunks (advance 64B per K-tile)
  uint64_t ga[6];
  ga[0] = (uint64_t)(uintptr_t)(A + (size_t)(bm0 + ar0) * K + ac0);
  ga[1] = (uint64_t)(uintptr_t)(A + (size_t)(bm0 + ar1) * K + ac1);
#pragma unroll
  for (int i = 0; i < 4; ++i) {
    const int id = t + 256 * i;
    ga[2 + i] = (uint64_t)(uintptr_t)(W + (size_t)(bn0 + (id >> 2)) * K + (id & 3) * 8);
  }
  // LDS destination byte offsets (buffer 0); buffer 1 adds the per-array delta.
  // (generic->LDS: flat address low 32 bits are the LDS byte address)
  unsigned la[6], ldelta[6];
  la[0] = (unsigned)(uintptr_t)(void*)&sA[0][ar0 * LDK + ac0];
  la[1] = (unsigned)(uintptr_t)(void*)&sA[0][ar1 * LDK + ac1];
  ldelta[0] = ldelta[1] = (unsigned)(BM * LDK * sizeof(bf16_t));
#pragma unroll
  for (int i = 0; i < 4; ++i) {
    const int id = t + 256 * i;
    la[2 + i] = (unsigned)(uintptr_t)(void*)&sB[0][(id >> 2) * LDK + (id & 3) * 8];
    ldelta[2 + i] = (unsigned)(BN * LDK * sizeof(bf16_t));
  }

  const int KT = K / BK;

  // prologue: stream tile 0 into buffer 0
#pragma unroll
  for (int i = 0; i < 6; ++i) {
    GFNet_async_b128(la[i], ga[i]);
    ga[i] += BK * sizeof(bf16_t);
  }

  for (int kt = 0; kt < KT; ++kt) {
    const int cur = kt & 1;
    if (kt + 1 < KT) {
      // stream tile kt+1 into the other buffer; overlaps WMMAs below
#pragma unroll
      for (int i = 0; i < 6; ++i) {
        GFNet_async_b128(la[i] + (cur ? 0u : ldelta[i]), ga[i]);
        ga[i] += BK * sizeof(bf16_t);
      }
      if (kt + 2 < KT) {  // warm L2 one tile further ahead
        __builtin_prefetch((const void*)(uintptr_t)ga[0], 0, 1);
        __builtin_prefetch((const void*)(uintptr_t)ga[2], 0, 1);
        __builtin_prefetch((const void*)(uintptr_t)ga[4], 0, 1);
      }
      asm volatile("s_wait_asynccnt 0x6" ::: "memory");  // tile kt landed
    } else {
      asm volatile("s_wait_asynccnt 0x0" ::: "memory");
    }
    __syncthreads();  // all waves' portions of tile kt visible in LDS

    // A fragment (16x32 bf16): lanes 0-15 hold K 0-7 / 16-23, lanes 16-31 K 8-15 / 24-31
    FragBF fa[4], fb[4];
#pragma unroll
    for (int mi = 0; mi < 4; ++mi) {
      const bf16_t* p = &sA[cur][(wm0 + mi * 16 + mrow) * LDK];
      fa[mi].q[0] = *(const uint4*)(p + half * 8);
      fa[mi].q[1] = *(const uint4*)(p + 16 + half * 8);
    }
    // B fragment via transposed layout [N][K]: lane n=mrow, lanes 0-15 K 0-7 / 16-23, lanes 16-31 K 8-15 / 24-31
#pragma unroll
    for (int ni = 0; ni < 4; ++ni) {
      const bf16_t* p = &sB[cur][(wn0 + ni * 16 + mrow) * LDK + half * 8];
      fb[ni].q[0] = *(const uint4*)(p);
      fb[ni].q[1] = *(const uint4*)(p + 16);
    }
#pragma unroll
    for (int mi = 0; mi < 4; ++mi)
#pragma unroll
      for (int ni = 0; ni < 4; ++ni)
        acc[mi][ni] = __builtin_amdgcn_wmma_f32_16x16x32_bf16(
            false, fa[mi].v, false, fb[ni].v, (short)0, acc[mi][ni], false, false);

    __syncthreads();  // reads done before this buffer is re-targeted (tile kt+2)
  }

  // Epilogue. C/D layout: vgpr j -> M = j + 8*(lane/16), N = lane%16.
#pragma unroll
  for (int mi = 0; mi < 4; ++mi)
#pragma unroll
    for (int ni = 0; ni < 4; ++ni)
#pragma unroll
      for (int j = 0; j < 8; ++j) {
        const int mm = bm0 + wm0 + mi * 16 + j + 8 * half;
        const int nn = bn0 + wn0 + ni * 16 + mrow;
        float v = acc[mi][ni][j];
        if (EPI == 0) {
          v += bias[nn] + pos[(mm & 255) * DIMC + nn];
          outF[(size_t)mm * N + nn] = v;
        } else if (EPI == 1) {
          const float h = v + bias[nn];
          outH[(size_t)mm * N + nn] =
              (bf16_t)(0.5f * h * (1.0f + erff(h * 0.70710678118654752f)));
        } else {
          float* p = outF + (size_t)mm * N + nn;
          *p = *p + v + bias[nn];
        }
      }
}

// ---------------------------------------------------------------------------
// LayerNorm: one wave32 per token (768 wide, 24 elems/lane).
// WF32 -> write f32, WH -> write bf16 (feeds next GEMM A operand).
// ---------------------------------------------------------------------------
template <int WF32, int WH>
__global__ __launch_bounds__(256)
void GFNet_ln(const float* __restrict__ src, const float* __restrict__ w,
              const float* __restrict__ b, float* __restrict__ dstF,
              bf16_t* __restrict__ dstH) {
  const int lane = threadIdx.x & 31;
  const int tok  = blockIdx.x * 8 + (threadIdx.x >> 5);
  const float* x = src + (size_t)tok * DIMC;

  float vals[24];
  float s = 0.0f;
#pragma unroll
  for (int i = 0; i < 24; ++i) { vals[i] = x[lane + 32 * i]; s += vals[i]; }
#pragma unroll
  for (int off = 16; off; off >>= 1) s += __shfl_xor(s, off, 32);
  const float mean = s * (1.0f / 768.0f);
  float vs = 0.0f;
#pragma unroll
  for (int i = 0; i < 24; ++i) { const float d = vals[i] - mean; vs += d * d; }
#pragma unroll
  for (int off = 16; off; off >>= 1) vs += __shfl_xor(vs, off, 32);
  const float inv = rsqrtf(vs * (1.0f / 768.0f) + 1e-6f);
#pragma unroll
  for (int i = 0; i < 24; ++i) {
    const int d = lane + 32 * i;
    const float o = (vals[i] - mean) * inv * w[d] + b[d];
    if (WF32) dstF[(size_t)tok * DIMC + d] = o;
    if (WH)   dstH[(size_t)tok * DIMC + d] = (bf16_t)o;
  }
}

// ---------------------------------------------------------------------------
// Global filter: per (batch, 16-channel group), 16x16 2D DFT -> complex
// modulation (Hermitian-expanded from the 16x9 rfft weights) -> inverse DFT.
// ortho fwd+inv == unnormalized fwd * W * unnormalized inv * (1/256).
// ---------------------------------------------------------------------------
__global__ __launch_bounds__(256)
void GFNet_gfilter(const float* __restrict__ src, const float* __restrict__ cwl,
                   float* __restrict__ dst) {
  __shared__ float2 tile[16][16][17];  // [h][w][d], pad to dodge bank conflicts
  const int t  = threadIdx.x;
  const int b  = blockIdx.x;
  const int d0 = blockIdx.y << 4;
  const int dd = t & 15;   // local channel
  const int aa = t >> 4;   // row/col the thread owns per phase

  float twr[16], twi[16];  // e^{-2*pi*i*j/16}
#pragma unroll
  for (int j = 0; j < 16; ++j) {
    float sn, cs;
    __sincosf(-0.39269908169872414f * (float)j, &sn, &cs);
    twr[j] = cs; twi[j] = sn;
  }

  const size_t base = (size_t)b * 256 * DIMC + d0 + dd;
#pragma unroll
  for (int h = 0; h < 16; ++h)
    tile[h][aa][dd] = make_float2(src[base + (size_t)(h * 16 + aa) * DIMC], 0.0f);
  __syncthreads();

  // forward DFT along w : thread owns (h=aa, d=dd)
  {
    float xr[16], xi[16];
#pragma unroll
    for (int w = 0; w < 16; ++w) { float2 v = tile[aa][w][dd]; xr[w] = v.x; xi[w] = v.y; }
#pragma unroll
    for (int k = 0; k < 16; ++k) {
      float sr = 0, si = 0;
#pragma unroll
      for (int w = 0; w < 16; ++w) {
        const int ix = (k * w) & 15;
        sr += xr[w] * twr[ix] - xi[w] * twi[ix];
        si += xr[w] * twi[ix] + xi[w] * twr[ix];
      }
      tile[aa][k][dd] = make_float2(sr, si);
    }
  }
  __syncthreads();

  // forward DFT along h : thread owns (kw=aa, d=dd)
  {
    float xr[16], xi[16];
#pragma unroll
    for (int h = 0; h < 16; ++h) { float2 v = tile[h][aa][dd]; xr[h] = v.x; xi[h] = v.y; }
#pragma unroll
    for (int k = 0; k < 16; ++k) {
      float sr = 0, si = 0;
#pragma unroll
      for (int h = 0; h < 16; ++h) {
        const int ix = (k * h) & 15;
        sr += xr[h] * twr[ix] - xi[h] * twi[ix];
        si += xr[h] * twi[ix] + xi[h] * twr[ix];
      }
      tile[k][aa][dd] = make_float2(sr, si);
    }
  }
  __syncthreads();

  // complex modulation : thread owns (kh=aa, kw=dd), loops channels
  {
    const int kh = aa, kw = dd;
#pragma unroll
    for (int d = 0; d < 16; ++d) {
      float cr, ci;
      if (kw <= 8) {
        const float* p = cwl + ((size_t)(kh * 9 + kw) * DIMC + d0 + d) * 2;
        cr = p[0]; ci = p[1];
      } else {  // Hermitian symmetry: W[kh,kw] = conj(W[(16-kh)%16, 16-kw])
        const float* p =
            cwl + ((size_t)(((16 - kh) & 15) * 9 + (16 - kw)) * DIMC + d0 + d) * 2;
        cr = p[0]; ci = -p[1];
      }
      const float2 v = tile[kh][kw][d];
      tile[kh][kw][d] = make_float2(v.x * cr - v.y * ci, v.x * ci + v.y * cr);
    }
  }
  __syncthreads();

  // inverse DFT along h (conj twiddles) : thread owns (kw=aa, d=dd)
  {
    float xr[16], xi[16];
#pragma unroll
    for (int k = 0; k < 16; ++k) { float2 v = tile[k][aa][dd]; xr[k] = v.x; xi[k] = v.y; }
#pragma unroll
    for (int h = 0; h < 16; ++h) {
      float sr = 0, si = 0;
#pragma unroll
      for (int k = 0; k < 16; ++k) {
        const int ix = (k * h) & 15;
        sr += xr[k] * twr[ix] + xi[k] * twi[ix];
        si += xi[k] * twr[ix] - xr[k] * twi[ix];
      }
      tile[h][aa][dd] = make_float2(sr, si);
    }
  }
  __syncthreads();

  // inverse DFT along w, keep real part, scale 1/256 : thread owns (h=aa, d=dd)
  {
    float xr[16], xi[16];
#pragma unroll
    for (int k = 0; k < 16; ++k) { float2 v = tile[aa][k][dd]; xr[k] = v.x; xi[k] = v.y; }
#pragma unroll
    for (int w = 0; w < 16; ++w) {
      float sr = 0;
#pragma unroll
      for (int k = 0; k < 16; ++k) {
        const int ix = (k * w) & 15;
        sr += xr[k] * twr[ix] + xi[k] * twi[ix];
      }
      dst[(size_t)(b * 256 + aa * 16 + w) * DIMC + d0 + dd] = sr * (1.0f / 256.0f);
    }
  }
}

// ---------------------------------------------------------------------------
// Small helpers
// ---------------------------------------------------------------------------
__global__ void GFNet_cvt_bf16(const float* __restrict__ s, bf16_t* __restrict__ d, int n) {
  const int i = blockIdx.x * 256 + threadIdx.x;
  if (i < n) d[i] = (bf16_t)s[i];
}

// im2col for the 16x16/stride16 patch conv: A[m][k], k = c*256 + ph*16 + pw
__global__ void GFNet_im2col(const float* __restrict__ x, bf16_t* __restrict__ a) {
  const int idx = blockIdx.x * 256 + threadIdx.x;        // < 16384*768
  const int m = idx / DIMC, k = idx - m * DIMC;
  const int b = m >> 8, tok = m & 255, hp = tok >> 4, wp = tok & 15;
  const int c = k >> 8, r = k & 255, ph = r >> 4, pw = r & 15;
  a[idx] = (bf16_t)x[(size_t)(((b * 3 + c) * 256 + hp * 16 + ph) * 256) + wp * 16 + pw];
}

__global__ void GFNet_meanpool(const float* __restrict__ src, float* __restrict__ feat) {
  const int d = blockIdx.y * 256 + threadIdx.x;  // grid (64,3) x 256
  const int b = blockIdx.x;
  float s = 0.0f;
  for (int tok = 0; tok < 256; ++tok) s += src[(size_t)(b * 256 + tok) * DIMC + d];
  feat[b * DIMC + d] = s * (1.0f / 256.0f);
}

__global__ void GFNet_head(const float* __restrict__ feat, const float* __restrict__ hw,
                           const float* __restrict__ hb, float* __restrict__ out) {
  const int b = blockIdx.x, o = threadIdx.x >> 5, lane = threadIdx.x & 31;
  float s = 0.0f;
  for (int i = lane; i < DIMC; i += 32) s += feat[b * DIMC + i] * hw[o * DIMC + i];
#pragma unroll
  for (int off = 16; off; off >>= 1) s += __shfl_xor(s, off, 32);
  if (lane == 0) out[b * 2 + o] = s + hb[o];
}

// ---------------------------------------------------------------------------
extern "C" void kernel_launch(void* const* d_in, const int* in_sizes, int n_in,
                              void* d_out, int out_size, void* d_ws, size_t ws_size,
                              hipStream_t stream) {
  (void)in_sizes; (void)n_in; (void)out_size; (void)ws_size;
  const float* x      = (const float*)d_in[0];
  const float* conv_w = (const float*)d_in[1];
  const float* conv_b = (const float*)d_in[2];
  const float* pos    = (const float*)d_in[3];
  const float* cw     = (const float*)d_in[4];
  const float* ln1_w  = (const float*)d_in[5];
  const float* ln1_b  = (const float*)d_in[6];
  const float* ln2_w  = (const float*)d_in[7];
  const float* ln2_b  = (const float*)d_in[8];
  const float* fc1_w  = (const float*)d_in[9];
  const float* fc1_b  = (const float*)d_in[10];
  const float* fc2_w  = (const float*)d_in[11];
  const float* fc2_b  = (const float*)d_in[12];
  const float* norm_w = (const float*)d_in[13];
  const float* norm_b = (const float*)d_in[14];
  const float* head_w = (const float*)d_in[15];
  const float* head_b = (const float*)d_in[16];
  float* out = (float*)d_out;

  char* ws = (char*)d_ws;
  size_t off = 0;
  auto wsalloc = [&](size_t bytes) {
    off = (off + 255) & ~(size_t)255;
    void* p = ws + off;
    off += bytes;
    return p;
  };
  float*  t        = (float*)wsalloc((size_t)NTOK * DIMC * 4);
  float*  f32a     = (float*)wsalloc((size_t)NTOK * DIMC * 4);   // ln1 / final-ln out
  float*  f32b     = (float*)wsalloc((size_t)NTOK * DIMC * 4);   // filter out
  bf16_t* lnh      = (bf16_t*)wsalloc((size_t)NTOK * DIMC * 2);  // ln2 out (GEMM A)
  bf16_t* hid      = (bf16_t*)wsalloc((size_t)NTOK * HIDC * 2);  // gelu(fc1) out
  bf16_t* apatch   = (bf16_t*)wsalloc((size_t)NTOK * DIMC * 2);  // im2col
  bf16_t* wb_conv  = (bf16_t*)wsalloc((size_t)DIMC * DIMC * 2);
  bf16_t* wb_fc1   = (bf16_t*)wsalloc((size_t)8 * HIDC * DIMC * 2);
  bf16_t* wb_fc2   = (bf16_t*)wsalloc((size_t)8 * DIMC * HIDC * 2);
  float*  feat     = (float*)wsalloc((size_t)64 * DIMC * 4);

  // weight conversion (bf16) + im2col
  {
    const int n1 = DIMC * DIMC, n2 = 8 * HIDC * DIMC;
    GFNet_cvt_bf16<<<(n1 + 255) / 256, 256, 0, stream>>>(conv_w, wb_conv, n1);
    GFNet_cvt_bf16<<<(n2 + 255) / 256, 256, 0, stream>>>(fc1_w, wb_fc1, n2);
    GFNet_cvt_bf16<<<(n2 + 255) / 256, 256, 0, stream>>>(fc2_w, wb_fc2, n2);
    GFNet_im2col<<<(NTOK * DIMC) / 256, 256, 0, stream>>>(x, apatch);
  }

  // patch embed: t = im2col(x) @ conv_w^T + conv_b + pos_embed
  GFNet_gemm<0><<<dim3(NTOK / BM, DIMC / BN), 256, 0, stream>>>(
      apatch, wb_conv, conv_b, pos, t, nullptr, NTOK, DIMC, DIMC);

  for (int i = 0; i < 8; ++i) {
    GFNet_ln<1, 0><<<NTOK / 8, 256, 0, stream>>>(
        t, ln1_w + i * DIMC, ln1_b + i * DIMC, f32a, nullptr);
    GFNet_gfilter<<<dim3(64, DIMC / 16), 256, 0, stream>>>(
        f32a, cw + (size_t)i * 16 * 9 * DIMC * 2, f32b);
    GFNet_ln<0, 1><<<NTOK / 8, 256, 0, stream>>>(
        f32b, ln2_w + i * DIMC, ln2_b + i * DIMC, nullptr, lnh);
    GFNet_gemm<1><<<dim3(NTOK / BM, HIDC / BN), 256, 0, stream>>>(
        lnh, wb_fc1 + (size_t)i * HIDC * DIMC, fc1_b + i * HIDC, nullptr,
        nullptr, hid, NTOK, HIDC, DIMC);
    GFNet_gemm<2><<<dim3(NTOK / BM, DIMC / BN), 256, 0, stream>>>(
        hid, wb_fc2 + (size_t)i * DIMC * HIDC, fc2_b + i * DIMC, nullptr,
        t, nullptr, NTOK, DIMC, HIDC);
  }

  GFNet_ln<1, 0><<<NTOK / 8, 256, 0, stream>>>(t, norm_w, norm_b, f32a, nullptr);
  GFNet_meanpool<<<dim3(64, DIMC / 256), 256, 0, stream>>>(f32a, feat);
  GFNet_head<<<64, 64, 0, stream>>>(feat, head_w, head_b, out);
}
